// biaffine_parser_79559974191114
// MI455X (gfx1250) — matrix operand order repaired and checked by reading, not compiled, not run on hardware
//
#include <hip/hip_runtime.h>

// Problem constants (match reference)
#define B_    8
#define L_    512
#define DH_   768
#define DARC_ 512
#define DREL_ 128
#define C_    37
#define BL_   (B_ * L_)

typedef __bf16 bf16_t;
typedef __attribute__((ext_vector_type(16))) __bf16 v16bf;
typedef __attribute__((ext_vector_type(8)))  __bf16 v8bf;
typedef __attribute__((ext_vector_type(8)))  float  v8f;
typedef __attribute__((ext_vector_type(4)))  unsigned int u32x4;
typedef __attribute__((ext_vector_type(8)))  int i32x8;
typedef __attribute__((ext_vector_type(4)))  int i32x4;

// ---------------------------------------------------------------------------
// Stage a contiguous run of `nelem` bf16 elements from global into LDS.
// Uses the Tensor Data Mover (1-D D#, ISA 8.3/8.4) when the builtin exists;
// wave issuing the TDM waits on TENSORcnt. Caller must __syncthreads() after.
// nelem must fit tile_dim0 (16 bits); all call sites are <= 12288.
// ---------------------------------------------------------------------------
__device__ __forceinline__ void stage_issue(bf16_t* lds, const bf16_t* gsrc,
                                            unsigned nelem, int tid, int nthreads) {
#if __has_builtin(__builtin_amdgcn_tensor_load_to_lds)
  if (tid == 0) {
    unsigned int lds_off = (unsigned int)(uintptr_t)(void*)lds;  // low 32b = LDS byte addr
    unsigned long long ga = (unsigned long long)(uintptr_t)gsrc;
    u32x4 g0;
    g0[0] = 1u;                                                    // count=1 (valid user D#)
    g0[1] = lds_off;                                               // lds_addr
    g0[2] = (unsigned int)(ga & 0xFFFFFFFFu);                      // global_addr[31:0]
    g0[3] = (unsigned int)((ga >> 32) & 0x01FFFFFFu) | (2u << 30); // addr[56:32] | type=2
    i32x8 g1;
    g1[0] = (int)(1u << 16);                             // wg_mask=0, data_size=1 (2B)
    g1[1] = (int)((nelem & 0xFFFFu) << 16);              // tensor_dim0[15:0]
    g1[2] = (int)(((nelem >> 16) & 0xFFFFu) | (1u << 16)); // tensor_dim0[31:16] | tensor_dim1=1
    g1[3] = (int)((nelem & 0xFFFFu) << 16);              // tensor_dim1[31:16]=0 | tile_dim0
    g1[4] = 1;                                           // tile_dim1=1, tile_dim2=0
    g1[5] = (int)nelem;                                  // tensor_dim0_stride
    g1[6] = 0;
    g1[7] = 0;
    i32x4 z4 = {0, 0, 0, 0};
#if defined(__clang_major__) && (__clang_major__ >= 23)
    i32x8 z8 = {0, 0, 0, 0, 0, 0, 0, 0};
    __builtin_amdgcn_tensor_load_to_lds(g0, g1, z4, z4, z8, 0);
#else
    __builtin_amdgcn_tensor_load_to_lds(g0, g1, z4, z4, 0);
#endif
    __builtin_amdgcn_s_wait_tensorcnt(0);
  }
#else
  const int nv = (int)(nelem * 2 / 16);
  for (int i = tid; i < nv; i += nthreads)
    ((uint4*)lds)[i] = ((const uint4*)gsrc)[i];
#endif
}

// ---------------------------------------------------------------------------
// Wave-level K-loop: acc += A(16xK) * B(Kx16), bf16->f32 WMMA.
// ap/bp must already include the per-lane fragment offsets:
//   ap = Abase + (lane&15)*lda + (lane&16 ? 16 : 0)
//   bp = Bbase + (lane&15)*ldb + (lane&16 ? 16 : 0)
// (contiguous k order, same for A and B: lane half h holds k = 16h..16h+15, wave32)
// ---------------------------------------------------------------------------
template <int K>
__device__ __forceinline__ v8f wmma_loop(const bf16_t* ap, const bf16_t* bp) {
  v8f acc = {};
#pragma unroll
  for (int k = 0; k < K; k += 32) {
    v8bf a0 = *(const v8bf*)(ap + k);
    v8bf a1 = *(const v8bf*)(ap + k + 8);
    v8bf b0 = *(const v8bf*)(bp + k);
    v8bf b1 = *(const v8bf*)(bp + k + 8);
    v16bf a, b;
#pragma unroll
    for (int i = 0; i < 8; ++i) {
      a[i] = a0[i]; a[i + 8] = a1[i];
      b[i] = b0[i]; b[i + 8] = b1[i];
    }
    acc = __builtin_amdgcn_wmma_f32_16x16x32_bf16(false, a, false, b,
                                                  (short)0, acc, false, false);
  }
  return acc;
}

// ---------------------------------------------------------------------------
// Small prep kernels
// ---------------------------------------------------------------------------
__global__ void k_f32_to_bf16(const float* __restrict__ x, bf16_t* __restrict__ y, int n) {
  int i = blockIdx.x * blockDim.x + threadIdx.x;
  if (i < n) y[i] = (bf16_t)x[i];
}

// Ut[c,e,d] = U[c,d,e]  (so B^T rows are K(=d)-contiguous)
__global__ void k_transpose_u(const float* __restrict__ U, bf16_t* __restrict__ Ut) {
  int i = blockIdx.x * blockDim.x + threadIdx.x;
  if (i >= C_ * DREL_ * DREL_) return;
  int c = i / (DREL_ * DREL_);
  int r = i % (DREL_ * DREL_);
  int d = r / DREL_;
  int e = r % DREL_;
  Ut[((size_t)c * DREL_ + e) * DREL_ + d] = (bf16_t)U[i];
}

// rb[i] = dot(Hah[i,:], bias_arc)
__global__ void k_rowdot(const bf16_t* __restrict__ Hahb, const float* __restrict__ bias_arc,
                         float* __restrict__ rb) {
  int i = blockIdx.x * blockDim.x + threadIdx.x;
  if (i >= BL_) return;
  float s = 0.0f;
  for (int k = 0; k < DARC_; ++k) s += (float)Hahb[(size_t)i * DARC_ + k] * bias_arc[k];
  rb[i] = s;
}

// aff[i,c] = bias_rel[c] + Hrh[i,:]·W_rel[c,:128] + Hrd[i,:]·W_rel[c,128:]
__global__ void k_affine(const bf16_t* __restrict__ Hrhb, const bf16_t* __restrict__ Hrdb,
                         const float* __restrict__ Wrel, const float* __restrict__ brel,
                         float* __restrict__ aff) {
  int idx = blockIdx.x * blockDim.x + threadIdx.x;
  if (idx >= BL_ * C_) return;
  int i = idx / C_;
  int c = idx % C_;
  const float* w = Wrel + (size_t)c * 2 * DREL_;
  float s = brel[c];
  for (int d = 0; d < DREL_; ++d)
    s += (float)Hrhb[(size_t)i * DREL_ + d] * w[d] +
         (float)Hrdb[(size_t)i * DREL_ + d] * w[DREL_ + d];
  aff[idx] = s;
}

// ---------------------------------------------------------------------------
// Generic GEMM: Cb[m,n] = (relu?)(A[m,:]·Wt[n,:] + bias[n])
// 8-wave block: one TDM-staged A tile (16xK in LDS) shared by 8 N-tiles.
// grid = (N/128, M/16), block = 256.
// ---------------------------------------------------------------------------
template <int K, bool RELU>
__global__ __launch_bounds__(256) void k_gemm_bias(const bf16_t* __restrict__ A,
                                                   const bf16_t* __restrict__ Wt,
                                                   const float* __restrict__ bias,
                                                   bf16_t* __restrict__ Cb, int N) {
  __shared__ __align__(16) bf16_t As[16 * K];
  int tid = threadIdx.x, lane = tid & 31, wave = tid >> 5;
  int mb = blockIdx.y * 16;
  int nb = blockIdx.x * 128 + wave * 16;

  stage_issue(As, A + (size_t)mb * K, 16u * K, tid, 256);
  __syncthreads();

  const bf16_t* ap = As + (lane & 15) * K + ((lane & 16) ? 16 : 0);
  const bf16_t* bp = Wt + (size_t)(nb + (lane & 15)) * K + ((lane & 16) ? 16 : 0);
  v8f acc = wmma_loop<K>(ap, bp);

  int h = lane >> 4, nl = lane & 15;
  float bv = bias ? bias[nb + nl] : 0.0f;
#pragma unroll
  for (int r = 0; r < 8; ++r) {
    int m = mb + r + 8 * h;
    float v = acc[r] + bv;
    if (RELU) v = fmaxf(v, 0.0f);
    Cb[(size_t)m * N + nb + nl] = (bf16_t)v;
  }
}

// ---------------------------------------------------------------------------
// S_arc[b,l,m] = T[b,l,:]·Hah[b,m,:] + rb[b,l]
// 8-wave block: TDM-staged T tile (16x512) shared by 8 m-tiles.
// grid = (L/128, L/16, B), block = 256.
// ---------------------------------------------------------------------------
__global__ __launch_bounds__(256) void k_arc(const bf16_t* __restrict__ Tb,
                                             const bf16_t* __restrict__ Hahb,
                                             const float* __restrict__ rb,
                                             float* __restrict__ Sarc) {
  __shared__ __align__(16) bf16_t As[16 * DARC_];
  int tid = threadIdx.x, lane = tid & 31, wave = tid >> 5;
  int mb = blockIdx.x * 128 + wave * 16;   // N dim (m)
  int lb = blockIdx.y * 16;                // M dim (l)
  int b  = blockIdx.z;

  stage_issue(As, Tb + (size_t)(b * L_ + lb) * DARC_, 16u * DARC_, tid, 256);
  __syncthreads();

  const bf16_t* ap = As + (lane & 15) * DARC_ + ((lane & 16) ? 16 : 0);
  const bf16_t* bp = Hahb + (size_t)(b * L_ + mb + (lane & 15)) * DARC_ + ((lane & 16) ? 16 : 0);
  v8f acc = wmma_loop<DARC_>(ap, bp);

  int h = lane >> 4, nl = lane & 15;
#pragma unroll
  for (int r = 0; r < 8; ++r) {
    int l = lb + r + 8 * h;
    Sarc[(size_t)(b * L_ + l) * L_ + mb + nl] = acc[r] + rb[b * L_ + l];
  }
}

// ---------------------------------------------------------------------------
// P[i,c,e] = Hrh[i,:]·Ut[c,e,:]   (bilinear stage 1)
// 8 waves = exactly the 8 e-tiles of one (i-tile, c); Hrh tile TDM-staged.
// grid = (BL/16, C), block = 256.
// ---------------------------------------------------------------------------
__global__ __launch_bounds__(256) void k_relp(const bf16_t* __restrict__ Hrhb,
                                              const bf16_t* __restrict__ Ut,
                                              bf16_t* __restrict__ P) {
  __shared__ __align__(16) bf16_t As[16 * DREL_];
  int tid = threadIdx.x, lane = tid & 31, wave = tid >> 5;
  int ib = blockIdx.x * 16;   // M dim (i = b*L+l)
  int c  = blockIdx.y;
  int eb = wave * 16;         // N dim (e)

  stage_issue(As, Hrhb + (size_t)ib * DREL_, 16u * DREL_, tid, 256);
  __syncthreads();

  const bf16_t* ap = As + (lane & 15) * DREL_ + ((lane & 16) ? 16 : 0);
  const bf16_t* bp = Ut + ((size_t)c * DREL_ + eb + (lane & 15)) * DREL_ + ((lane & 16) ? 16 : 0);
  v8f acc = wmma_loop<DREL_>(ap, bp);

  int h = lane >> 4, nl = lane & 15;
#pragma unroll
  for (int r = 0; r < 8; ++r) {
    int i = ib + r + 8 * h;
    P[((size_t)i * C_ + c) * DREL_ + eb + nl] = (bf16_t)acc[r];
  }
}

// ---------------------------------------------------------------------------
// Bilinear stage 2 + epilogue. One 256-thread block owns ALL 37 label planes
// of one (b, l-tile, m-tile):
//   - Hrd 16x128 tile TDM-staged into LDS, B fragments hoisted into registers
//     (invariant across c) -> each label GEMM = 8 A-loads + 4 WMMAs
//   - aff 16x37 tile staged into LDS (contiguous run)
//   - 16x16x37 output tile accumulated in LDS, then cooperatively stored
//     fully coalesced (each l-row = 592 contiguous floats)
// grid = (L/16, L/16, B), block = 256.
// ---------------------------------------------------------------------------
__global__ __launch_bounds__(256) void k_relout(const bf16_t* __restrict__ P,
                                                const bf16_t* __restrict__ Hrdb,
                                                const float* __restrict__ aff,
                                                float* __restrict__ Srel) {
  __shared__ __align__(16) bf16_t Bs[16 * DREL_];    // Hrd tile [m][k], 4KB
  __shared__ float AffS[16 * C_];                    // aff tile [l][c], 2368B
  __shared__ float Obuf[16 * 16 * C_];               // [l][m][c], 37888B

  int tid  = threadIdx.x;
  int lane = tid & 31;
  int wave = tid >> 5;
  int mb = blockIdx.x * 16;
  int lb = blockIdx.y * 16;
  int b  = blockIdx.z;

  stage_issue(Bs, Hrdb + (size_t)(b * L_ + mb) * DREL_, 16u * DREL_, tid, 256);
  for (int i = tid; i < 16 * C_; i += 256)
    AffS[i] = aff[(size_t)(b * L_ + lb) * C_ + i];
  __syncthreads();

  int h = lane >> 4, nl = lane & 15;

  // Hoist B fragments into registers: invariant across all 37 labels.
  const bf16_t* bp = &Bs[(lane & 15) * DREL_ + ((lane & 16) ? 16 : 0)];
  v16bf bfrag[4];
#pragma unroll
  for (int kk = 0; kk < 4; ++kk) {
    v8bf b0 = *(const v8bf*)(bp + kk * 32);
    v8bf b1 = *(const v8bf*)(bp + kk * 32 + 8);
#pragma unroll
    for (int i = 0; i < 8; ++i) { bfrag[kk][i] = b0[i]; bfrag[kk][i + 8] = b1[i]; }
  }

  for (int c = wave; c < C_; c += 8) {
    const bf16_t* ap = P + ((size_t)(b * L_ + lb + (lane & 15)) * C_ + c) * DREL_ +
                       ((lane & 16) ? 16 : 0);
    v8f acc = {};
#pragma unroll
    for (int kk = 0; kk < 4; ++kk) {
      v8bf a0 = *(const v8bf*)(ap + kk * 32);
      v8bf a1 = *(const v8bf*)(ap + kk * 32 + 8);
      v16bf a;
#pragma unroll
      for (int i = 0; i < 8; ++i) { a[i] = a0[i]; a[i + 8] = a1[i]; }
      acc = __builtin_amdgcn_wmma_f32_16x16x32_bf16(false, a, false, bfrag[kk],
                                                    (short)0, acc, false, false);
    }
#pragma unroll
    for (int r = 0; r < 8; ++r) {
      int l = r + 8 * h;
      Obuf[(l * 16 + nl) * C_ + c] = acc[r] + AffS[l * C_ + c];
    }
  }
  __syncthreads();

  // Coalesced tile store: Obuf layout (l*16+m)*37+c matches output inner order.
  const int TOT = 16 * 16 * C_;  // 9472
  for (int idx = tid; idx < TOT; idx += 256) {
    int l   = idx / (16 * C_);
    int rem = idx - l * (16 * C_);
    Srel[((size_t)(b * L_ + lb + l) * L_ + mb) * C_ + rem] = Obuf[idx];
  }
}

// ---------------------------------------------------------------------------
// Host launcher
// ---------------------------------------------------------------------------
extern "C" void kernel_launch(void* const* d_in, const int* in_sizes, int n_in,
                              void* d_out, int out_size, void* d_ws, size_t ws_size,
                              hipStream_t stream) {
  const float* H        = (const float*)d_in[0];
  const float* W_ah     = (const float*)d_in[1];
  const float* b_ah     = (const float*)d_in[2];
  const float* W_ad     = (const float*)d_in[3];
  const float* b_ad     = (const float*)d_in[4];
  const float* W_arc    = (const float*)d_in[5];
  const float* bias_arc = (const float*)d_in[6];
  const float* W_rh     = (const float*)d_in[7];
  const float* b_rh     = (const float*)d_in[8];
  const float* W_rd     = (const float*)d_in[9];
  const float* b_rd     = (const float*)d_in[10];
  const float* U_rel    = (const float*)d_in[11];
  const float* W_rel    = (const float*)d_in[12];
  const float* bias_rel = (const float*)d_in[13];

  char* ws = (char*)d_ws;
  size_t off = 0;
  auto alloc = [&](size_t bytes) -> char* {
    char* p = ws + off;
    off += (bytes + 255) & ~(size_t)255;
    return p;
  };

  bf16_t* Hb    = (bf16_t*)alloc((size_t)BL_ * DH_ * 2);
  bf16_t* Wahb  = (bf16_t*)alloc((size_t)DARC_ * DH_ * 2);
  bf16_t* Wadb  = (bf16_t*)alloc((size_t)DARC_ * DH_ * 2);
  bf16_t* Wrhb  = (bf16_t*)alloc((size_t)DREL_ * DH_ * 2);
  bf16_t* Wrdb  = (bf16_t*)alloc((size_t)DREL_ * DH_ * 2);
  bf16_t* Warcb = (bf16_t*)alloc((size_t)DARC_ * DARC_ * 2);
  bf16_t* Ut    = (bf16_t*)alloc((size_t)C_ * DREL_ * DREL_ * 2);
  bf16_t* Hahb  = (bf16_t*)alloc((size_t)BL_ * DARC_ * 2);
  bf16_t* Hadb  = (bf16_t*)alloc((size_t)BL_ * DARC_ * 2);
  bf16_t* Hrhb  = (bf16_t*)alloc((size_t)BL_ * DREL_ * 2);
  bf16_t* Hrdb  = (bf16_t*)alloc((size_t)BL_ * DREL_ * 2);
  bf16_t* Tb    = (bf16_t*)alloc((size_t)BL_ * DARC_ * 2);
  float*  rb    = (float*)alloc((size_t)BL_ * 4);
  bf16_t* P     = (bf16_t*)alloc((size_t)BL_ * C_ * DREL_ * 2);
  float*  aff   = (float*)alloc((size_t)BL_ * C_ * 4);
  (void)ws_size; (void)in_sizes; (void)n_in; (void)out_size;

  float* Sarc = (float*)d_out;
  float* Srel = Sarc + (size_t)B_ * L_ * L_;

  auto cvt = [&](const float* src, bf16_t* dst, int n) {
    k_f32_to_bf16<<<(n + 255) / 256, 256, 0, stream>>>(src, dst, n);
  };

  // 1) precision conversion + U transpose
  cvt(H, Hb, BL_ * DH_);
  cvt(W_ah, Wahb, DARC_ * DH_);
  cvt(W_ad, Wadb, DARC_ * DH_);
  cvt(W_rh, Wrhb, DREL_ * DH_);
  cvt(W_rd, Wrdb, DREL_ * DH_);
  cvt(W_arc, Warcb, DARC_ * DARC_);
  k_transpose_u<<<(C_ * DREL_ * DREL_ + 255) / 256, 256, 0, stream>>>(U_rel, Ut);

  // 2) four MLP GEMMs (bias + ReLU); 8 N-tiles share one LDS A tile
  k_gemm_bias<DH_, true><<<dim3(DARC_ / 128, BL_ / 16), 256, 0, stream>>>(Hb, Wahb, b_ah, Hahb, DARC_);
  k_gemm_bias<DH_, true><<<dim3(DARC_ / 128, BL_ / 16), 256, 0, stream>>>(Hb, Wadb, b_ad, Hadb, DARC_);
  k_gemm_bias<DH_, true><<<dim3(DREL_ / 128, BL_ / 16), 256, 0, stream>>>(Hb, Wrhb, b_rh, Hrhb, DREL_);
  k_gemm_bias<DH_, true><<<dim3(DREL_ / 128, BL_ / 16), 256, 0, stream>>>(Hb, Wrdb, b_rd, Hrdb, DREL_);

  // 3) T = Had @ W_arc^T  (no bias/relu)
  k_gemm_bias<DARC_, false><<<dim3(DARC_ / 128, BL_ / 16), 256, 0, stream>>>(Hadb, Warcb, nullptr, Tb, DARC_);

  // 4) arc path: row term + batched S_arc GEMM
  k_rowdot<<<(BL_ + 255) / 256, 256, 0, stream>>>(Hahb, bias_arc, rb);
  k_arc<<<dim3(L_ / 128, L_ / 16, B_), 256, 0, stream>>>(Tb, Hahb, rb, Sarc);

  // 5) rel path: bilinear stage 1, affine term, bilinear stage 2 + epilogue
  k_relp<<<dim3(BL_ / 16, C_), 256, 0, stream>>>(Hrhb, Ut, P);
  k_affine<<<(BL_ * C_ + 255) / 256, 256, 0, stream>>>(Hrhb, Hrdb, W_rel, bias_rel, aff);
  k_relout<<<dim3(L_ / 16, L_ / 16, B_), 256, 0, stream>>>(P, Hrdb, aff, Srel);
}
